// EchoStateNetwork_20804821581983
// MI455X (gfx1250) — matrix operand order repaired and checked by reading, not compiled, not run on hardware
//
#include <hip/hip_runtime.h>
#include <hip/hip_bf16.h>

// ---------------------------------------------------------------------------
// Echo State Network on MI455X (gfx1250, wave32, WMMA bf16 16x16x32).
//
// Decomposition: 32 persistent workgroups = 2 batch-groups (16 rows = 1 WMMA
// M-tile) x 16 column-groups (64 cols). 4 waves/WG, one 16x16 tile per wave.
// W_res slice (64x1024 bf16 = 128KB) + W_in slice live in LDS for the whole
// run; per step only h (bf16 A-fragments) moves through L2. Global step
// barrier = monotonic atomic counter in workspace. tanh via hardware
// V_TANH_F32; output states streamed with non-temporal stores.
// ---------------------------------------------------------------------------

typedef __attribute__((ext_vector_type(16))) __bf16 v16bf;
typedef __attribute__((ext_vector_type(8)))  float  v8f;

#define B_   32
#define T_   4096
#define D_   64
#define N_   1024
#define LEAK 0.9f

// workspace layout (bytes)
#define BAR_OFF    0
#define HBUF_OFF   1024
#define HBUF_BYTES (2*2*32*32*16*2)              // 2 buf x 2 bg x 32 kk x 32 lane x 16 bf16 = 128KB
#define WRES_OFF   (HBUF_OFF + HBUF_BYTES)       // 2MB packed W_res fragments
#define WRES_BYTES (N_*N_*2)
#define WIN_OFF    (WRES_OFF + WRES_BYTES)       // 128KB packed W_in fragments
#define WIN_BYTES  (N_*64*2)
#define XF_OFF     (WIN_OFF + WIN_BYTES)         // 16MB packed x A-fragments
#define XF_ELEMS   (2*T_*2*32*16)                // 8,388,608 bf16

__device__ __forceinline__ float fast_tanh(float v) {
#if __has_builtin(__builtin_amdgcn_tanh_f32)
    return __builtin_amdgcn_tanh_f32(v);         // gfx1250 V_TANH_F32 (single trans op)
#elif __has_builtin(__builtin_amdgcn_tanhf)
    return __builtin_amdgcn_tanhf(v);
#else
    return tanhf(v);
#endif
}

// ---- pre-pass: zero barrier counter + h buffers -----------------------------
__global__ void esn_zero_k(unsigned* __restrict__ p, int nwords) {
    int i = blockIdx.x * 256 + threadIdx.x;
    if (i < nwords) p[i] = 0u;
}

// ---- pre-pass: W_res -> bf16 B-fragment layout [n_tile][kk][lane][idx] ------
// B 32x16 bf16 fragment: lane = N + ((k_sub&8)?16:0);
// idx = (k_sub&7) | ((k_sub&16)>>1), for column N = lane&15.   (ISA 7.12.2)
__global__ void esn_pack_wres_k(const float* __restrict__ wres, __bf16* __restrict__ dst) {
    int i = blockIdx.x * 256 + threadIdx.x;      // 64*32*32*16 = 1,048,576
    int idx  =  i        & 15;
    int lane = (i >> 4)  & 31;
    int kk   = (i >> 9)  & 31;
    int nt   =  i >> 14;
    int n = nt * 16 + (lane & 15);
    int k = kk * 32 + ((lane >> 4) << 3) + (idx & 7) + ((idx & 8) << 1);
    dst[i] = (__bf16)wres[n * N_ + k];
}

// ---- pre-pass: W_in[:, :64] -> bf16 B-fragments [n_tile][kk(2)][lane][idx] --
__global__ void esn_pack_win_k(const float* __restrict__ win, __bf16* __restrict__ dst) {
    int i = blockIdx.x * 256 + threadIdx.x;      // 64*2*32*16 = 65,536
    int idx  =  i        & 15;
    int lane = (i >> 4)  & 31;
    int kk   = (i >> 9)  & 1;
    int nt   =  i >> 10;
    int n = nt * 16 + (lane & 15);
    int k = kk * 32 + ((lane >> 4) << 3) + (idx & 7) + ((idx & 8) << 1);
    dst[i] = (__bf16)win[n * (D_ + 1) + k];
}

// ---- pre-pass: input x -> bf16 A-fragment layout [bg][t][c(2)][lane][idx] ---
// A 16x32 bf16 fragment: lane = M + ((k_sub&8)?16:0);
// idx = (k_sub&7) | ((k_sub&16)>>1).              (ISA 7.12.2)
__global__ void esn_pack_x_k(const float* __restrict__ x, __bf16* __restrict__ dst) {
    int i = blockIdx.x * 256 + threadIdx.x;      // 2*4096*2*32*16 = 8,388,608
    int idx  =  i        & 15;
    int lane = (i >> 4)  & 31;
    int c    = (i >> 9)  & 1;
    int t    = (i >> 10) & 4095;
    int bg   = (i >> 22) & 1;
    int hi = lane >> 4;
    int ksub = (idx & 7) | (hi << 3) | ((idx & 8) << 1);
    int k = c * 32 + ksub;
    int b = bg * 16 + (lane & 15);
    dst[i] = (__bf16)x[((size_t)b * T_ + t) * D_ + k];
}

// ---- main persistent recurrence kernel --------------------------------------
__global__ __launch_bounds__(128)
void esn_main_k(const float*  __restrict__ win_raw,     // for bias column
                const __bf16* __restrict__ wres_pack,
                const __bf16* __restrict__ win_pack,
                const __bf16* __restrict__ xfrag,
                __bf16*       __restrict__ hbuf,
                unsigned*     __restrict__ bar,
                float*        __restrict__ out) {
    // 128KB W_res slice + 8KB W_in slice + bias: fits 320KB WGP LDS
    __shared__ __align__(32) __bf16 s_wres[4][32][32][16];
    __shared__ __align__(32) __bf16 s_win [4][2][32][16];
    __shared__ float s_bias[64];

    const int tid  = threadIdx.x;
    const int w    = tid >> 5;          // wave -> 16-col sub-tile
    const int lane = tid & 31;
    const int bg   = blockIdx.x >> 4;   // batch group (16 rows)
    const int cg   = blockIdx.x & 15;   // column group (64 cols)

    // --- stage weights into LDS (once) ---
    {
        uint4*       d  = (uint4*)&s_wres[0][0][0][0];
        const uint4* s  = (const uint4*)wres_pack + (size_t)cg * 8192;   // 128KB slab
        for (int i = tid; i < 8192; i += 128) d[i] = s[i];
        uint4*       d2 = (uint4*)&s_win[0][0][0][0];
        const uint4* s2 = (const uint4*)win_pack + (size_t)cg * 512;     // 8KB slab
        for (int i = tid; i < 512; i += 128) d2[i] = s2[i];
        for (int i = tid; i < 64; i += 128)
            s_bias[i] = win_raw[(size_t)(cg * 64 + i) * (D_ + 1) + D_];
    }
    __syncthreads();

    // fp32 h state lives in registers: thread (lane,r) owns (b_loc, n_loc)
    float hreg[8];
#pragma unroll
    for (int r = 0; r < 8; ++r) hreg[r] = 0.0f;

    const int   n_loc = lane & 15;
    const int   b_off = (lane >> 4) << 3;           // C layout: lanes 16-31 -> M+8
    const int   n_g   = cg * 64 + w * 16 + n_loc;   // global column == next-step K index
    const float bias  = s_bias[w * 16 + n_loc];
    const int   ksub  = n_g & 31;
    const int   kk2   = n_g >> 5;
    const int   laneh = (ksub & 8) << 1;            // +16 lane offset in A fragment
    const int   idx2  = (ksub & 7) | ((ksub & 16) >> 1);

    for (int t = 0; t < T_; ++t) {
        const int cur = t & 1, nxt = cur ^ 1;
        v8f acc = {0.f, 0.f, 0.f, 0.f, 0.f, 0.f, 0.f, 0.f};

        // h(t-1) @ W_res^T : 32 K-chunks of 32 (bf16 WMMA), A from global (L2-hot)
        const __bf16* hsrc = hbuf + (size_t)(cur * 2 + bg) * 16384;
#pragma unroll 4
        for (int kk = 0; kk < 32; ++kk) {
            v16bf a = *(const v16bf*)(hsrc + (kk * 32 + lane) * 16);
            v16bf b = *(const v16bf*)(&s_wres[w][kk][lane][0]);
            acc = __builtin_amdgcn_wmma_f32_16x16x32_bf16(
                      false, a, false, b, (short)0, acc, false, false);
        }
        // + x_t @ W_in^T : 2 more K-chunks (fused input drive)
        const __bf16* xsrc = xfrag + (size_t)(bg * T_ + t) * 1024;
#pragma unroll
        for (int c = 0; c < 2; ++c) {
            v16bf a = *(const v16bf*)(xsrc + (c * 32 + lane) * 16);
            v16bf b = *(const v16bf*)(&s_win[w][c][lane][0]);
            acc = __builtin_amdgcn_wmma_f32_16x16x32_bf16(
                      false, a, false, b, (short)0, acc, false, false);
        }

        // epilogue: bias + hw tanh + leak; publish fp32 states (non-temporal,
        // write-once 512MB stream) and bf16 h(t) fragments for next step
        __bf16* hdst = hbuf + (size_t)(nxt * 2 + bg) * 16384;
#pragma unroll
        for (int r = 0; r < 8; ++r) {
            const int b_loc = r + b_off;
            float hn = fast_tanh(acc[r] + bias);
            float h  = (1.0f - LEAK) * hreg[r] + LEAK * hn;
            hreg[r] = h;
            const int b = bg * 16 + b_loc;
            __builtin_nontemporal_store(h, &out[((size_t)b * T_ + t) * N_ + n_g]);
            hdst[(kk2 * 32 + (b_loc | laneh)) * 16 + idx2] = (__bf16)h;
        }

        // device-scope step barrier (monotonic counter, 32 blocks)
        __threadfence();
        __syncthreads();
        if (tid == 0) {
            atomicAdd(bar, 1u);
            const unsigned target = 32u * (unsigned)(t + 1);
            while (__hip_atomic_load(bar, __ATOMIC_RELAXED, __HIP_MEMORY_SCOPE_AGENT) < target) {
                __builtin_amdgcn_s_sleep(1);
            }
        }
        __syncthreads();
        __threadfence();
    }
}

// ---------------------------------------------------------------------------
extern "C" void kernel_launch(void* const* d_in, const int* in_sizes, int n_in,
                              void* d_out, int out_size, void* d_ws, size_t ws_size,
                              hipStream_t stream) {
    const float* x    = (const float*)d_in[0];   // [32,4096,64]
    const float* win  = (const float*)d_in[1];   // [1024,65]
    const float* wres = (const float*)d_in[2];   // [1024,1024]
    float* out = (float*)d_out;                  // [32,4096,1024] fp32

    char* ws = (char*)d_ws;                      // needs ~18.3 MB
    unsigned* bar       = (unsigned*)(ws + BAR_OFF);
    __bf16*   hbuf      = (__bf16*)(ws + HBUF_OFF);
    __bf16*   wres_pack = (__bf16*)(ws + WRES_OFF);
    __bf16*   win_pack  = (__bf16*)(ws + WIN_OFF);
    __bf16*   xfrag     = (__bf16*)(ws + XF_OFF);

    // zero barrier counter + both h double-buffers (h0 = 0)
    const int zwords = (HBUF_OFF + HBUF_BYTES) / 4;
    esn_zero_k<<<(zwords + 255) / 256, 256, 0, stream>>>((unsigned*)ws, zwords);
    esn_pack_wres_k<<<1048576 / 256, 256, 0, stream>>>(wres, wres_pack);
    esn_pack_win_k <<<65536  / 256, 256, 0, stream>>>(win, win_pack);
    esn_pack_x_k   <<<XF_ELEMS / 256, 256, 0, stream>>>(x, xfrag);
    esn_main_k<<<32, 128, 0, stream>>>(win, wres_pack, win_pack, xfrag, hbuf, bar, out);
}
